// TimeAwareEncoderBlock_84404697301444
// MI455X (gfx1250) — matrix operand order repaired and checked by reading, not compiled, not run on hardware
//
#include <hip/hip_runtime.h>
#include <hip/hip_bf16.h>

// ---------------------------------------------------------------------------
// TimeAwareEncoderBlock for MI455X (gfx1250, wave32, WMMA f16 16x16x32).
// B=2 S=2048 D=1024 H=16 G=4 HD=64 E=8 K=2 F=4096, T=4096.
// f16 WMMA for all big GEMMs; MoE fused per 16-token tile (hidden [16x4096]
// f16 in LDS, expert weights L2-resident); flash attention with LDS-staged
// transposed K and V tiles; async global->LDS copies for tile staging.
// ---------------------------------------------------------------------------

#define Tn   4096
#define Dn   1024
#define Hn   16
#define Gn   4
#define HDn  64
#define En   8
#define Fn   4096
#define Sn   2048
#define QKVN 1536   // 1024 q + 256 k + 256 v columns

typedef __attribute__((ext_vector_type(16))) _Float16 v16h;
typedef __attribute__((ext_vector_type(8)))  _Float16 v8h;
typedef __attribute__((ext_vector_type(8)))  float    v8f;

__device__ __forceinline__ v8f wmma16(v16h a, v16h b, v8f c) {
  // D = A(16x32 f16) * B(32x16 f16) + C(16x16 f32)
  return __builtin_amdgcn_wmma_f32_16x16x32_f16(false, a, false, b, (short)0, c, false, false);
}

// A-fragment (16x32, f16) from a row-major f16 buffer (LDS).
// lane L: M = L&15 ; half j: K = (j&7) + 8*(L>>4) + 16*(j>>3)   (ISA 7.12.2)
__device__ __forceinline__ v16h load_a_frag(const _Float16* base, int ld, int k0, int lane) {
  int m = lane & 15, hi = lane >> 4;
  v16h a{};
#pragma unroll
  for (int j = 0; j < 16; ++j) {
    int k = (j & 7) + 8 * hi + 16 * (j >> 3);
    a[j] = base[m * ld + k0 + k];
  }
  return a;
}

// Async global -> LDS copy of one aligned 16B chunk (ASYNCcnt-tracked DMA
// path; CDNA5 GLOBAL_LOAD_ASYNC_TO_LDS_B128, ISA 10.7/15.18).
__device__ __forceinline__ void async_b128(_Float16* lds_dst, const _Float16* gsrc) {
  unsigned ldsoff = (unsigned)(unsigned long long)lds_dst;  // LDS aperture: addr[31:0] = offset
  asm volatile("global_load_async_to_lds_b128 %0, %1, off"
               :: "v"(ldsoff), "v"((unsigned long long)gsrc)
               : "memory");
}
__device__ __forceinline__ void async_wait0() {
  asm volatile("s_wait_asynccnt 0" ::: "memory");
}

// ---------------------------------------------------------------------------
// small prep kernels
// ---------------------------------------------------------------------------
__global__ void f32_to_f16_kernel(const float* __restrict__ src, _Float16* __restrict__ dst, long n) {
  long i = (long)blockIdx.x * blockDim.x + threadIdx.x;
  if (i < n) dst[i] = (_Float16)src[i];
}

__global__ void pack_qkv_kernel(const float* __restrict__ wq, const float* __restrict__ wk,
                                const float* __restrict__ wv, const float* __restrict__ bq,
                                const float* __restrict__ bk, const float* __restrict__ bv,
                                _Float16* __restrict__ wqkv, float* __restrict__ bias) {
  int i = blockIdx.x * blockDim.x + threadIdx.x;  // over D*1536
  if (i >= Dn * QKVN) return;
  int d = i / QKVN, j = i % QKVN;
  float w = (j < 1024) ? wq[d * 1024 + j]
          : (j < 1280) ? wk[d * 256 + (j - 1024)]
                       : wv[d * 256 + (j - 1280)];
  wqkv[i] = (_Float16)w;
  if (d == 0)
    bias[j] = (j < 1024) ? bq[j] : (j < 1280) ? bk[j - 1024] : bv[j - 1280];
}

__global__ void layernorm_f16_kernel(const float* __restrict__ x, const float* __restrict__ scale,
                                     const float* __restrict__ bias, _Float16* __restrict__ out) {
  int t = blockIdx.x;
  const float* row = x + (long)t * Dn;
  __shared__ float red[256];
  float s = 0.f;
  for (int i = threadIdx.x; i < Dn; i += 256) s += row[i];
  red[threadIdx.x] = s; __syncthreads();
  for (int off = 128; off > 0; off >>= 1) {
    if (threadIdx.x < off) red[threadIdx.x] += red[threadIdx.x + off];
    __syncthreads();
  }
  float mu = red[0] / (float)Dn; __syncthreads();
  float v = 0.f;
  for (int i = threadIdx.x; i < Dn; i += 256) { float d = row[i] - mu; v += d * d; }
  red[threadIdx.x] = v; __syncthreads();
  for (int off = 128; off > 0; off >>= 1) {
    if (threadIdx.x < off) red[threadIdx.x] += red[threadIdx.x + off];
    __syncthreads();
  }
  float rstd = rsqrtf(red[0] / (float)Dn + 1e-6f);
  for (int i = threadIdx.x; i < Dn; i += 256)
    out[(long)t * Dn + i] = (_Float16)((row[i] - mu) * rstd * scale[i] + bias[i]);
}

// ---------------------------------------------------------------------------
// generic WMMA GEMM: per block, 16 rows x 512 cols. Each wave owns 64 cols
// (4 WMMA tiles); one A fragment feeds 4 WMMAs; B pointer strength-reduced.
// ---------------------------------------------------------------------------
__global__ void gemm_f16_kernel(const _Float16* __restrict__ A, int lda,
                                const _Float16* __restrict__ Bm, int ldb,
                                const float* __restrict__ bias,
                                const float* __restrict__ residual,
                                _Float16* __restrict__ out_h, float* __restrict__ out_f,
                                int ldc, int Kdim) {
  extern __shared__ _Float16 AsG[];  // 16 x Kdim
  int mt = blockIdx.x, nb = blockIdx.y;
  int tid = threadIdx.x, lane = tid & 31, wv = tid >> 5;
  const _Float16* Arow = A + (long)mt * 16 * lda;
  // async-stage the 16xK A tile into LDS in 16B chunks
  int nchunk = 16 * Kdim / 8;
  for (int ci = tid; ci < nchunk; ci += 256) {
    int el = ci * 8, r = el / Kdim, c = el - r * Kdim;
    async_b128(&AsG[el], &Arow[(long)r * lda + c]);
  }
  async_wait0();
  __syncthreads();

  int n_base = nb * 512 + wv * 64;
  v8f acc[4] = {{}, {}, {}, {}};
  const _Float16* brow = Bm + n_base + (lane & 15);  // per-lane B base
  long bstep = (long)32 * ldb;
  for (int kk = 0; kk < Kdim; kk += 32, brow += bstep) {
    v16h a = load_a_frag(AsG, Kdim, kk, lane);
    v16h b0{}, b1{}, b2{}, b3{};
#pragma unroll
    for (int j = 0; j < 16; ++j) {
      const _Float16* p = brow + (long)((j & 7) + 8 * (lane >> 4) + 16 * (j >> 3)) * ldb;
      b0[j] = p[0]; b1[j] = p[16]; b2[j] = p[32]; b3[j] = p[48];
    }
    acc[0] = wmma16(a, b0, acc[0]);
    acc[1] = wmma16(a, b1, acc[1]);
    acc[2] = wmma16(a, b2, acc[2]);
    acc[3] = wmma16(a, b3, acc[3]);
  }
  int hi = lane >> 4;
#pragma unroll
  for (int j = 0; j < 4; ++j) {
    int ncol = n_base + j * 16 + (lane & 15);
    float bv = bias ? bias[ncol] : 0.f;
#pragma unroll
    for (int r = 0; r < 8; ++r) {
      int m = mt * 16 + r + 8 * hi;
      float v = acc[j][r] + bv;
      if (out_f) out_f[(long)m * ldc + ncol] = v + (residual ? residual[(long)m * ldc + ncol] : 0.f);
      else       out_h[(long)m * ldc + ncol] = (_Float16)v;
    }
  }
}

// ---------------------------------------------------------------------------
// flash attention with temporal-decay bias. grid = B*H*(S/128), 8 waves/block,
// each wave owns a 16-query tile. K tile staged transposed in LDS, V tile
// async-staged row-major; both shared by all 8 waves per 32-key step.
// ---------------------------------------------------------------------------
__global__ void attn_kernel(const _Float16* __restrict__ qkv, const float* __restrict__ deltas,
                            const float* __restrict__ time_decay, _Float16* __restrict__ o_h) {
  int bid = blockIdx.x;
  int qt = bid & 15;           // S/128 tiles
  int h  = (bid >> 4) & 15;
  int bb = bid >> 8;
  int tid = threadIdx.x, lane = tid & 31, wv = tid >> 5;
  int g = h >> 2;              // GQA group
  int q0 = qt * 128 + wv * 16;
  int m = lane & 15, hi = lane >> 4;

  __shared__ _Float16 KT[64 * 32];     // [hd][key] transposed, 4KB
  __shared__ _Float16 VT[32 * 64];     // [key][hd] row-major, 4KB
  __shared__ _Float16 PL[8][16 * 32];  // per-wave P tile scratch, 8KB

  // Q fragments over HD=64 (two K=32 steps)
  v16h a0{}, a1{};
#pragma unroll
  for (int j = 0; j < 16; ++j) {
    int k = (j & 7) + 8 * hi + 16 * (j >> 3);
    long t = (long)(bb * Sn + q0 + m);
    a0[j] = qkv[t * QKVN + h * 64 + k];
    a1[j] = qkv[t * QKVN + h * 64 + 32 + k];
  }
  float sp; { float td = time_decay[h]; sp = (td > 20.f) ? td : log1pf(expf(td)); }
  float dq[8];
#pragma unroll
  for (int r = 0; r < 8; ++r) dq[r] = deltas[bb * Sn + q0 + r + 8 * hi];

  float mrow[8], lrow[8];
  v8f O[4] = {{}, {}, {}, {}};
#pragma unroll
  for (int r = 0; r < 8; ++r) { mrow[r] = -1e30f; lrow[r] = 0.f; }

  int key = tid >> 3, hd0 = (tid & 7) * 8;  // staging assignment: 8 halves/thread

  for (int kb = 0; kb < Sn; kb += 32) {
    // ---- cooperative K/V staging (one 32-key step for all 8 waves) ----
    long ts = (long)(bb * Sn + kb + key);
    v8h kv = *(const v8h*)(qkv + ts * QKVN + 1024 + g * 64 + hd0);
#pragma unroll
    for (int jj = 0; jj < 8; ++jj) KT[(hd0 + jj) * 32 + key] = kv[jj];  // transpose
    async_b128(&VT[key * 64 + hd0], qkv + ts * QKVN + 1280 + g * 64 + hd0);
    async_wait0();
    __syncthreads();

    // ---- scores: s[16q x 32k] as two C fragments ----
    v8f s[2];
#pragma unroll
    for (int c = 0; c < 2; ++c) {
      v16h b0{}, b1{};
#pragma unroll
      for (int j = 0; j < 16; ++j) {
        int k = (j & 7) + 8 * hi + 16 * (j >> 3);
        b0[j] = KT[k * 32 + c * 16 + m];         // hd rows 0..31
        b1[j] = KT[(32 + k) * 32 + c * 16 + m];  // hd rows 32..63
      }
      v8f z = {};
      z = wmma16(a0, b0, z);
      z = wmma16(a1, b1, z);
      s[c] = z;
    }
    float dk0 = deltas[bb * Sn + kb + m];
    float dk1 = deltas[bb * Sn + kb + 16 + m];
    float scale_r[8];
#pragma unroll
    for (int r = 0; r < 8; ++r) {
      s[0][r] = s[0][r] * 0.125f - sp * fabsf(dq[r] - dk0);
      s[1][r] = s[1][r] * 0.125f - sp * fabsf(dq[r] - dk1);
      float mx = fmaxf(s[0][r], s[1][r]);
      for (int off = 8; off > 0; off >>= 1) mx = fmaxf(mx, __shfl_xor(mx, off, 32));
      float mn = fmaxf(mrow[r], mx);
      float p0 = expf(s[0][r] - mn), p1 = expf(s[1][r] - mn);
      float rs = p0 + p1;
      for (int off = 8; off > 0; off >>= 1) rs += __shfl_xor(rs, off, 32);
      float al = expf(mrow[r] - mn);
      lrow[r] = lrow[r] * al + rs;
      mrow[r] = mn;
      scale_r[r] = al;
      PL[wv][(r + 8 * hi) * 32 + m]      = (_Float16)p0;
      PL[wv][(r + 8 * hi) * 32 + 16 + m] = (_Float16)p1;
    }
#pragma unroll
    for (int nt = 0; nt < 4; ++nt)
#pragma unroll
      for (int r = 0; r < 8; ++r) O[nt][r] *= scale_r[r];
    __syncthreads();
    // ---- P.V accumulation ----
    v16h pa = load_a_frag(&PL[wv][0], 32, 0, lane);  // C-layout -> A-layout via LDS
#pragma unroll
    for (int nt = 0; nt < 4; ++nt) {
      v16h bv{};
#pragma unroll
      for (int j = 0; j < 16; ++j)
        bv[j] = VT[((j & 7) + 8 * hi + 16 * (j >> 3)) * 64 + nt * 16 + m];  // K-dim row = key, col = hd
      O[nt] = wmma16(pa, bv, O[nt]);
    }
    __syncthreads();  // protect KT/VT before restage
  }
#pragma unroll
  for (int nt = 0; nt < 4; ++nt)
#pragma unroll
    for (int r = 0; r < 8; ++r) {
      int qq = q0 + r + 8 * hi;
      float val = O[nt][r] / lrow[r];
      o_h[((long)(bb * Sn + qq)) * Dn + h * 64 + nt * 16 + m] = (_Float16)val;
    }
}

// ---------------------------------------------------------------------------
// router: logits + softmax + top2 + expert bucket append
// ---------------------------------------------------------------------------
__global__ void zero_cnt_kernel(int* cnt) { if (threadIdx.x < En) cnt[threadIdx.x] = 0; }

__global__ void router_kernel(const _Float16* __restrict__ xn, const float* __restrict__ rw,
                              const float* __restrict__ rb, int* __restrict__ cnt,
                              int* __restrict__ eidx, float* __restrict__ ewt) {
  int t = blockIdx.x;
  __shared__ float part[256 * En];
  float acc[En];
#pragma unroll
  for (int e = 0; e < En; ++e) acc[e] = 0.f;
  for (int d = threadIdx.x; d < Dn; d += 256) {
    float xv = (float)xn[(long)t * Dn + d];
#pragma unroll
    for (int e = 0; e < En; ++e) acc[e] += xv * rw[d * En + e];
  }
#pragma unroll
  for (int e = 0; e < En; ++e) part[threadIdx.x * En + e] = acc[e];
  __syncthreads();
  for (int off = 128; off > 0; off >>= 1) {
    if (threadIdx.x < off)
#pragma unroll
      for (int e = 0; e < En; ++e) part[threadIdx.x * En + e] += part[(threadIdx.x + off) * En + e];
    __syncthreads();
  }
  if (threadIdx.x == 0) {
    float lg[En], mx = -1e30f;
    for (int e = 0; e < En; ++e) { lg[e] = part[e] + rb[e]; mx = fmaxf(mx, lg[e]); }
    float sum = 0.f;
    for (int e = 0; e < En; ++e) { lg[e] = expf(lg[e] - mx); sum += lg[e]; }
    int i0 = 0;
    for (int e = 1; e < En; ++e) if (lg[e] > lg[i0]) i0 = e;
    int i1 = (i0 == 0) ? 1 : 0;
    for (int e = 0; e < En; ++e) if (e != i0 && lg[e] > lg[i1]) i1 = e;
    float v0 = lg[i0], v1 = lg[i1], inv = 1.f / (v0 + v1);
    int s0 = atomicAdd(&cnt[i0], 1); eidx[i0 * Tn + s0] = t; ewt[i0 * Tn + s0] = v0 * inv;
    int s1 = atomicAdd(&cnt[i1], 1); eidx[i1 * Tn + s1] = t; ewt[i1 * Tn + s1] = v1 * inv;
  }
}

// ---------------------------------------------------------------------------
// fused MoE: per 16-token tile of one expert. Gathered A async-staged in LDS
// (32KB), hidden [16x4096] f16 in LDS (128KB). GEMM1+GeLU -> LDS, GEMM2 ->
// atomicAdd into output. 4-wide N blocking; B pointers strength-reduced.
// ---------------------------------------------------------------------------
__global__ void moe_kernel(const _Float16* __restrict__ xn, const _Float16* __restrict__ w1,
                           const float* __restrict__ b1, const _Float16* __restrict__ w2,
                           const float* __restrict__ b2, const int* __restrict__ cnt,
                           const int* __restrict__ eidx, const float* __restrict__ ewt,
                           float* __restrict__ out) {
  extern __shared__ char lds[];
  int*       tok = (int*)lds;                              // 16 ints
  float*     gw  = (float*)(lds + 64);                     // 16 floats
  _Float16*  As  = (_Float16*)(lds + 128);                 // 16*1024 halves
  _Float16*  Hs  = (_Float16*)(lds + 128 + 16 * Dn * 2);   // 16*4096 halves

  int e = blockIdx.x >> 8;        // 256 tiles per expert
  int tile = blockIdx.x & 255;
  int n_e = cnt[e];
  if (tile * 16 >= n_e) return;   // uniform
  int tid = threadIdx.x, lane = tid & 31, wv = tid >> 5;

  if (tid < 16) {
    int r = tile * 16 + tid;
    if (r < n_e) { tok[tid] = eidx[e * Tn + r]; gw[tid] = ewt[e * Tn + r]; }
    else         { tok[tid] = 0;                gw[tid] = 0.f; }
  }
  __syncthreads();
  // async gather-stage: 16 rows x 1024 cols in 16B chunks (8 chunks/thread)
  for (int ci = tid; ci < 16 * Dn / 8; ci += 256) {
    int el = ci * 8, r = el >> 10, c = el & (Dn - 1);
    async_b128(&As[el], &xn[(long)tok[r] * Dn + c]);
  }
  async_wait0();
  __syncthreads();

  const _Float16* W1 = w1 + (long)e * Dn * Fn;
  for (int gb = 0; gb < 8; ++gb) {                  // 8 groups x 4 tiles x 16 = 512 cols/wave
    int n_base = wv * 512 + gb * 64;
    v8f acc[4] = {{}, {}, {}, {}};
    const _Float16* brow = W1 + n_base + (lane & 15);
    for (int kk = 0; kk < Dn; kk += 32, brow += (long)32 * Fn) {
      v16h a = load_a_frag(As, Dn, kk, lane);
      v16h b0{}, b1{}, b2{}, b3{};
#pragma unroll
      for (int j = 0; j < 16; ++j) {
        const _Float16* p = brow + (long)((j & 7) + 8 * (lane >> 4) + 16 * (j >> 3)) * Fn;
        b0[j] = p[0]; b1[j] = p[16]; b2[j] = p[32]; b3[j] = p[48];
      }
      acc[0] = wmma16(a, b0, acc[0]);
      acc[1] = wmma16(a, b1, acc[1]);
      acc[2] = wmma16(a, b2, acc[2]);
      acc[3] = wmma16(a, b3, acc[3]);
    }
    int hi = lane >> 4;
#pragma unroll
    for (int j = 0; j < 4; ++j) {
      int ncol = n_base + j * 16 + (lane & 15);
      float bias = b1[e * Fn + ncol];
#pragma unroll
      for (int r = 0; r < 8; ++r) {
        float xv = acc[j][r] + bias;
        float gl = 0.5f * xv * (1.f + tanhf(0.7978845608f * (xv + 0.044715f * xv * xv * xv)));
        Hs[(r + 8 * hi) * Fn + ncol] = (_Float16)gl;
      }
    }
  }
  __syncthreads();

  const _Float16* W2 = w2 + (long)e * Fn * Dn;
  for (int gb = 0; gb < 2; ++gb) {                  // 2 groups x 4 tiles x 16 = 128 cols/wave
    int n_base = wv * 128 + gb * 64;
    v8f acc[4] = {{}, {}, {}, {}};
    const _Float16* brow = W2 + n_base + (lane & 15);
    for (int kk = 0; kk < Fn; kk += 32, brow += (long)32 * Dn) {
      v16h a = load_a_frag(Hs, Fn, kk, lane);
      v16h b0{}, b1{}, b2{}, b3{};
#pragma unroll
      for (int j = 0; j < 16; ++j) {
        const _Float16* p = brow + (long)((j & 7) + 8 * (lane >> 4) + 16 * (j >> 3)) * Dn;
        b0[j] = p[0]; b1[j] = p[16]; b2[j] = p[32]; b3[j] = p[48];
      }
      acc[0] = wmma16(a, b0, acc[0]);
      acc[1] = wmma16(a, b1, acc[1]);
      acc[2] = wmma16(a, b2, acc[2]);
      acc[3] = wmma16(a, b3, acc[3]);
    }
    int hi = lane >> 4;
#pragma unroll
    for (int j = 0; j < 4; ++j) {
      int ncol = n_base + j * 16 + (lane & 15);
      float bias = b2[e * Dn + ncol];
#pragma unroll
      for (int r = 0; r < 8; ++r) {
        int mm = r + 8 * hi;
        float y = (acc[j][r] + bias) * gw[mm];
        atomicAdd(&out[(long)tok[mm] * Dn + ncol], y);
      }
    }
  }
}

// ---------------------------------------------------------------------------
extern "C" void kernel_launch(void* const* d_in, const int* in_sizes, int n_in,
                              void* d_out, int out_size, void* d_ws, size_t ws_size,
                              hipStream_t stream) {
  (void)in_sizes; (void)n_in; (void)out_size; (void)ws_size;
  const float* x        = (const float*)d_in[0];
  const float* deltas   = (const float*)d_in[1];
  const float* ln1_s    = (const float*)d_in[2];
  const float* ln1_b    = (const float*)d_in[3];
  const float* wq       = (const float*)d_in[4];
  const float* bq       = (const float*)d_in[5];
  const float* wk       = (const float*)d_in[6];
  const float* bk       = (const float*)d_in[7];
  const float* wvv      = (const float*)d_in[8];
  const float* bv       = (const float*)d_in[9];
  const float* wo       = (const float*)d_in[10];
  const float* bo       = (const float*)d_in[11];
  const float* tdec     = (const float*)d_in[12];
  const float* ln2_s    = (const float*)d_in[13];
  const float* ln2_b    = (const float*)d_in[14];
  const float* rw       = (const float*)d_in[15];
  const float* rb       = (const float*)d_in[16];
  const float* w1       = (const float*)d_in[17];
  const float* b1       = (const float*)d_in[18];
  const float* w2       = (const float*)d_in[19];
  const float* b2       = (const float*)d_in[20];
  float* out = (float*)d_out;

  char* ws = (char*)d_ws;
  size_t off = 0;
  auto alloc = [&](size_t bytes) { size_t p = off; off += (bytes + 255) & ~(size_t)255; return p; };
  _Float16* wqkv_h = (_Float16*)(ws + alloc((size_t)Dn * QKVN * 2));
  float*    bqkv   = (float*)   (ws + alloc((size_t)QKVN * 4));
  _Float16* wo_h   = (_Float16*)(ws + alloc((size_t)Dn * Dn * 2));
  _Float16* w1_h   = (_Float16*)(ws + alloc((size_t)En * Dn * Fn * 2));
  _Float16* w2_h   = (_Float16*)(ws + alloc((size_t)En * Fn * Dn * 2));
  _Float16* xn_h   = (_Float16*)(ws + alloc((size_t)Tn * Dn * 2));
  _Float16* qkv_h  = (_Float16*)(ws + alloc((size_t)Tn * QKVN * 2));
  _Float16* o_h    = (_Float16*)(ws + alloc((size_t)Tn * Dn * 2));
  _Float16* x1n_h  = (_Float16*)(ws + alloc((size_t)Tn * Dn * 2));
  int*      cnt    = (int*)     (ws + alloc((size_t)En * 4));
  int*      eidx   = (int*)     (ws + alloc((size_t)En * Tn * 4));
  float*    ewt    = (float*)   (ws + alloc((size_t)En * Tn * 4));

  // --- weight precision conversion (f32 -> f16, each call; deterministic)
  pack_qkv_kernel<<<(Dn * QKVN + 255) / 256, 256, 0, stream>>>(wq, wk, wvv, bq, bk, bv, wqkv_h, bqkv);
  f32_to_f16_kernel<<<((long)Dn * Dn + 255) / 256, 256, 0, stream>>>(wo, wo_h, (long)Dn * Dn);
  f32_to_f16_kernel<<<((long)En * Dn * Fn + 255) / 256, 256, 0, stream>>>(w1, w1_h, (long)En * Dn * Fn);
  f32_to_f16_kernel<<<((long)En * Fn * Dn + 255) / 256, 256, 0, stream>>>(w2, w2_h, (long)En * Fn * Dn);

  // --- LN1 ---
  layernorm_f16_kernel<<<Tn, 256, 0, stream>>>(x, ln1_s, ln1_b, xn_h);

  // --- fused QKV projection: [4096,1024] @ [1024,1536] ---
  gemm_f16_kernel<<<dim3(Tn / 16, QKVN / 512), 256, 16 * Dn * 2, stream>>>(
      xn_h, Dn, wqkv_h, QKVN, bqkv, nullptr, qkv_h, nullptr, QKVN, Dn);

  // --- flash attention with temporal decay ---
  attn_kernel<<<2 * Hn * (Sn / 128), 256, 0, stream>>>(qkv_h, deltas, tdec, o_h);

  // --- output projection + residual: writes x1 = x + attn_out into d_out ---
  gemm_f16_kernel<<<dim3(Tn / 16, Dn / 512), 256, 16 * Dn * 2, stream>>>(
      o_h, Dn, wo_h, Dn, bo, x, nullptr, out, Dn, Dn);

  // --- LN2 (reads x1 from d_out) ---
  layernorm_f16_kernel<<<Tn, 256, 0, stream>>>(out, ln2_s, ln2_b, x1n_h);

  // --- router + expert bucketing ---
  zero_cnt_kernel<<<1, 32, 0, stream>>>(cnt);
  router_kernel<<<Tn, 256, 0, stream>>>(x1n_h, rw, rb, cnt, eidx, ewt);

  // --- fused MoE (GEMM1 + GeLU + GEMM2), accumulates into d_out ---
  size_t moe_lds = 128 + (size_t)16 * Dn * 2 + (size_t)16 * Fn * 2;  // 160.1 KB
  moe_kernel<<<En * (Tn / 16), 256, moe_lds, stream>>>(
      x1n_h, w1_h, b1, w2_h, b2, cnt, eidx, ewt, out);
}
